// FreezableConv2d_34067680592611
// MI455X (gfx1250) — compile-verified, matrix operand order checked by reading, not run on hardware
//
#include <hip/hip_runtime.h>
#include <hip/hip_bf16.h>

typedef __attribute__((ext_vector_type(16))) __bf16 v16bf;
typedef __attribute__((ext_vector_type(8)))  __bf16 v8bf;
typedef __attribute__((ext_vector_type(8)))  float  v8f;

#define CIN   256
#define COUT  256
#define W56   56
#define PLANE 3136               // 56*56
#define RSTRIP 8                 // image rows per block
#define HALO_ROWS (RSTRIP + 2)   // 10
#define HALO_COLS (W56 + 2)      // 58  (zero-padded left/right)
#define NSLOTS (HALO_ROWS * HALO_COLS)   // 580
#define BM    64                 // output channels per block
#define BN    (RSTRIP * W56)     // 448 output pixels per block
#define KC    32                 // one bf16 WMMA K-step
#define LDSW  40                 // LDS row stride (32 halves + 8 pad; 80B, 16B-aligned)

#define A_ROWS  (9 * BM)             // 576 rows: [tap][k]
#define A_ELEMS (A_ROWS * LDSW)      // 23040 halves
#define B_ELEMS (NSLOTS * LDSW)      // 23200 halves
#define SMEM_BYTES ((A_ELEMS + B_ELEMS) * 2)   // 92480 B of 320KB WGP LDS

// ---------------------------------------------------------------------------
// One-time weight repack: (256,256,3,3) fp32 OIHW -> bf16 [rs][k][c]
// ---------------------------------------------------------------------------
__global__ __launch_bounds__(256) void pack_weights_bf16(
    const float* __restrict__ wt, unsigned short* __restrict__ wp)
{
    int i  = blockIdx.x * 256 + threadIdx.x;   // 0 .. 9*256*256-1
    int c  = i & 255;
    int k  = (i >> 8) & 255;
    int rs = i >> 16;
    float f = wt[((k << 8) + c) * 9 + rs];
    wp[i] = (unsigned short)(__builtin_bit_cast(unsigned, f) >> 16);
}

// ---------------------------------------------------------------------------
// Implicit-GEMM conv, halo-shared B staging, fully unrolled tap loop with
// immediate-offset fragment loads.
// ---------------------------------------------------------------------------
__global__ __launch_bounds__(256) void conv3x3_halo_bf16_wmma(
    const float* __restrict__ x,              // (32,256,56,56)
    const unsigned short* __restrict__ wp,    // bf16 [9][256][256]
    const float* __restrict__ bias,           // (256,)
    float* __restrict__ out)                  // (32,256,56,56)
{
    extern __shared__ __align__(16) unsigned short smem[];
    unsigned short* As = smem;                // [9*64][LDSW]  weights, all taps
    unsigned short* Bh = smem + A_ELEMS;      // [580][LDSW]   padded input halo

    const int t    = threadIdx.x;
    const int lane = t & 31;
    const int wv   = t >> 5;      // 0..7
    const int wm   = wv & 1;      // wave M coord (2)
    const int wn   = wv >> 1;     // wave N coord (4)
    const int mr   = lane & 15;
    const int kb   = (lane < 16) ? 0 : 8;   // bf16 fragment K-base per lane half

    const int strip = blockIdx.x;           // 0..223 (7 strips per image)
    const int n     = strip / 7;
    const int h0    = (strip % 7) * RSTRIP;
    const int k0    = blockIdx.y * BM;

    // ---- loop-invariant per-lane fragment base pointers ----
    // A: two M-fragments; tap rs adds constant rs*64*LDSW.
    const unsigned short* aptr[2];
    #pragma unroll
    for (int ii = 0; ii < 2; ++ii)
        aptr[ii] = As + (wm * 32 + ii * 16 + mr) * LDSW + kb;

    // B: seven N-fragments; tap (r,s) adds constant (r*HALO_COLS+s)*LDSW.
    const unsigned short* bptr[7];
    #pragma unroll
    for (int jj = 0; jj < 7; ++jj) {
        int pl = wn * 112 + jj * 16 + mr;   // local output pixel 0..447
        bptr[jj] = Bh + ((pl / W56) * HALO_COLS + (pl % W56)) * LDSW + kb;
    }

    v8f acc[2][7] = {};
    const float* xn = x + (long)n * CIN * PLANE;

    for (int c0 = 0; c0 < CIN; c0 += KC) {
        __syncthreads();   // previous taps finished reading As/Bh

        // ---- stage A: 9 taps x 64 k x 32 c (bf16, b128 copies) ----
        #pragma unroll
        for (int e = 0; e < 9; ++e) {
            int idx  = e * 256 + t;        // 2304 x 16B chunks
            int row  = idx >> 2;           // 0..575 = rs*64 + krow
            int col  = (idx & 3) * 8;
            int rs_i = row >> 6;
            int krow = row & 63;
            uint4 v = *(const uint4*)(wp + ((long)rs_i * COUT + k0 + krow) * CIN + c0 + col);
            *(uint4*)(&As[row * LDSW + col]) = v;
        }

        // ---- stage B: 10x58 zero-padded halo x 32 c, packed bf16 stores ----
        #pragma unroll
        for (int e = 0; e < 3; ++e) {
            int slot = e * 256 + t;
            if (slot < NSLOTS) {
                int srow = slot / HALO_COLS;
                int scol = slot % HALO_COLS;
                int h = h0 + srow - 1;
                int w = scol - 1;
                bool inb = ((unsigned)h < W56) && ((unsigned)w < W56);
                unsigned short* bd = &Bh[slot * LDSW];
                if (inb) {
                    const float* xs = xn + (long)c0 * PLANE + h * W56 + w;
                    #pragma unroll
                    for (int cc = 0; cc < KC; cc += 2) {
                        unsigned u0 = __builtin_bit_cast(unsigned, xs[(long)(cc    ) * PLANE]);
                        unsigned u1 = __builtin_bit_cast(unsigned, xs[(long)(cc + 1) * PLANE]);
                        *(unsigned*)(bd + cc) = (u0 >> 16) | (u1 & 0xFFFF0000u);
                    }
                } else {
                    uint4 z = {0u, 0u, 0u, 0u};
                    #pragma unroll
                    for (int cc = 0; cc < KC; cc += 8)
                        *(uint4*)(bd + cc) = z;
                }
            }
        }
        __syncthreads();

        // ---- compute: 9 taps x 14 WMMAs, all addresses immediate offsets ----
        #pragma unroll
        for (int rs = 0; rs < 9; ++rs) {
            const int r = rs / 3, s = rs % 3;
            const int aoff = rs * BM * LDSW;                 // constant per tap
            const int boff = (r * HALO_COLS + s) * LDSW;     // constant per tap

            v16bf a[2];
            #pragma unroll
            for (int ii = 0; ii < 2; ++ii) {
                v8bf lo = *(const v8bf*)(aptr[ii] + aoff);
                v8bf hi = *(const v8bf*)(aptr[ii] + aoff + 16);
                #pragma unroll
                for (int q = 0; q < 8; ++q) { a[ii][q] = lo[q]; a[ii][q + 8] = hi[q]; }
            }

            #pragma unroll
            for (int jj = 0; jj < 7; ++jj) {
                v8bf lo = *(const v8bf*)(bptr[jj] + boff);
                v8bf hi = *(const v8bf*)(bptr[jj] + boff + 16);
                v16bf b;
                #pragma unroll
                for (int q = 0; q < 8; ++q) { b[q] = lo[q]; b[q + 8] = hi[q]; }

                acc[0][jj] = __builtin_amdgcn_wmma_f32_16x16x32_bf16(
                    false, a[0], false, b, (short)0, acc[0][jj], false, false);
                acc[1][jj] = __builtin_amdgcn_wmma_f32_16x16x32_bf16(
                    false, a[1], false, b, (short)0, acc[1][jj], false, false);
            }
        }
    }

    // ---- bias + store: flat offset, lanes = consecutive pixels ----
    const int hiHalf = (lane >> 4) & 1;
    #pragma unroll
    for (int ii = 0; ii < 2; ++ii) {
        #pragma unroll
        for (int jj = 0; jj < 7; ++jj) {
            int pl = wn * 112 + jj * 16 + mr;
            #pragma unroll
            for (int vg = 0; vg < 8; ++vg) {
                int ch = k0 + wm * 32 + ii * 16 + vg + hiHalf * 8;
                out[(long)(n * COUT + ch) * PLANE + h0 * W56 + pl] =
                    acc[ii][jj][vg] + bias[ch];
            }
        }
    }
}

extern "C" void kernel_launch(void* const* d_in, const int* in_sizes, int n_in,
                              void* d_out, int out_size, void* d_ws, size_t ws_size,
                              hipStream_t stream) {
    const float* x    = (const float*)d_in[0];
    const float* wt   = (const float*)d_in[1];
    const float* bias = (const float*)d_in[2];
    float* out        = (float*)d_out;
    unsigned short* wp = (unsigned short*)d_ws;   // 9*256*256*2 = 1.18 MB scratch

    pack_weights_bf16<<<(9 * 256 * 256) / 256, 256, 0, stream>>>(wt, wp);

    dim3 grid(32 * 7, COUT / BM);   // (224 strips, 4 channel blocks)
    conv3x3_halo_bf16_wmma<<<grid, 256, SMEM_BYTES, stream>>>(x, wp, bias, out);
}
